// PerformerMultiHeadBlock_12962211299726
// MI455X (gfx1250) — matrix operand order repaired and checked by reading, not compiled, not run on hardware
//
#include <hip/hip_runtime.h>
#include <hip/hip_bf16.h>

// ---------------------------------------------------------------------------
// Performer multi-head attention block on gfx1250 (MI455X), bf16 WMMA path.
// B=2, N=16384, C=512, H=8, D=64, M=266 (padded to MP=288 = 9*32).
//  - all GEMMs via v_wmma_f32_16x16x32_bf16 (wave32, 16x16 D tile)
//  - kp stored TRANSPOSED [MP,N] straight out of the phi D-fragment
//    (lane holds 8 consecutive rows of one column -> contiguous stores),
//    so the context GEMM reads row-major and ksum is a coalesced reduce
//  - register blocking: 16x64 per wave; all 4 B fragments loaded BEFORE the
//    4 WMMAs each K-step so loads overlap the matrix pipe (partial waits)
//  - A panels for the C=512 GEMMs staged via global_load_async_to_lds_b128
// ---------------------------------------------------------------------------

typedef __bf16 bf16_t;
typedef __attribute__((ext_vector_type(16))) __bf16 v16bf;
typedef __attribute__((ext_vector_type(8)))  float  v8f;

#define Bc   2
#define Nc   16384
#define Cc   512
#define Hc   8
#define Dc   64
#define Mc   266
#define MPc  288
#define BHc  (Bc * Hc)

#define KERNEL_EPS 1e-3f
#define LN_EPS     1e-5f

// ---- fragment helpers (layouts per CDNA5 ISA §7.12.2) ----------

__device__ inline v16bf load_a_row(const bf16_t* A, int lda) {
  int lane  = threadIdx.x & 31;
  int khalf = (lane < 16) ? 0 : 8;
  const bf16_t* row = A + (size_t)(lane & 15) * lda;
  v16bf a;
#pragma unroll
  for (int i = 0; i < 8; ++i) {
    int k = (i < 4) ? (2 * i + khalf) : (2 * i + 8 + khalf);
    a[2 * i]     = row[k];
    a[2 * i + 1] = row[k + 1];
  }
  return a;
}

__device__ inline v16bf load_b_row(const bf16_t* Bm, int ldb) {
  int lane = threadIdx.x & 31;
  int khalf = (lane < 16) ? 0 : 8;
  const bf16_t* col = Bm + (lane & 15);
  v16bf b;
#pragma unroll
  for (int i = 0; i < 16; ++i) {
    int k = (i < 8) ? (i + khalf) : (i + 8 + khalf);
    b[i] = col[(size_t)k * ldb];
  }
  return b;
}

__device__ inline v8f wmma_bf16(v16bf a, v16bf b, v8f c) {
  return __builtin_amdgcn_wmma_f32_16x16x32_bf16(
      false, a, false, b, (short)0, c, false, false);
}

// Stage a contiguous 16x512 bf16 panel (16 KB) global->LDS with the CDNA5
// async-to-LDS path.  1024 chunks of 16B; 128 threads x 8 chunks each.
__device__ inline void stage_panel_async(const bf16_t* __restrict__ gsrc,
                                         bf16_t* lds, int tid) {
  unsigned lbase = (unsigned)(uintptr_t)lds;   // low 32 bits = LDS byte addr
#pragma unroll
  for (int c = 0; c < 8; ++c) {
    int chunk = c * 128 + tid;                 // 0..1023
    asm volatile("global_load_async_to_lds_b128 %0, %1, off"
                 :: "v"(lbase + chunk * 16), "v"(gsrc + chunk * 8)
                 : "memory");
  }
  asm volatile("s_wait_asynccnt 0x0" ::: "memory");
  __syncthreads();
}

// ---------------------------------------------------------------------------
// K0: pack/convert inputs to bf16, build proj^T padded.
// ---------------------------------------------------------------------------
#define PK_H  ((size_t)Bc * Nc * Cc)        // 16,777,216
#define PK_W  ((size_t)Cc * Cc)             // 262,144
#define PK_P  ((size_t)Dc * MPc)            // 18,432
#define PK_TOTAL (PK_H + 4 * PK_W + PK_P)

__global__ void k_pack(const float* __restrict__ h,
                       const float* __restrict__ Wq, const float* __restrict__ Wk,
                       const float* __restrict__ Wv, const float* __restrict__ Wo,
                       const float* __restrict__ proj,
                       bf16_t* __restrict__ hb, bf16_t* __restrict__ wb,
                       bf16_t* __restrict__ projb) {
  size_t i = (size_t)blockIdx.x * blockDim.x + threadIdx.x;
  if (i < PK_H) { hb[i] = (bf16_t)h[i]; return; }
  i -= PK_H;
  if (i < 4 * PK_W) {
    const float* W = (i < PK_W) ? Wq : (i < 2 * PK_W) ? Wk : (i < 3 * PK_W) ? Wv : Wo;
    wb[i] = (bf16_t)W[i % PK_W];
    return;
  }
  i -= 4 * PK_W;
  if (i < PK_P) {                      // projb[d][m] = proj[m][d], zero-padded
    int d = (int)(i / MPc), m = (int)(i % MPc);
    projb[i] = (m < Mc) ? (bf16_t)proj[(size_t)m * Dc + d] : (bf16_t)0.0f;
  }
}

// ---------------------------------------------------------------------------
// K1: q/k/v = hb @ W{q,k,v} + b -> head layout [B,H,N,D] bf16.
// grid (2048*2, 1, 3), block 128. Block: 16 rows x 256 cols; A panel in LDS.
// ---------------------------------------------------------------------------
__global__ void k_qkv(const bf16_t* __restrict__ hb, const bf16_t* __restrict__ wb,
                      const float* __restrict__ bq, const float* __restrict__ bk,
                      const float* __restrict__ bv, bf16_t* __restrict__ qkv) {
  __shared__ bf16_t As[16 * Cc];               // 16 KB
  int lane = threadIdx.x & 31, wid = threadIdx.x >> 5;
  int tm = blockIdx.x >> 1;                    // row tile over B*N/16
  int cg = blockIdx.x & 1;                     // column group (256 cols)
  int z  = blockIdx.z;
  const bf16_t* W    = wb + (size_t)z * Cc * Cc;
  const float*  bias = (z == 0) ? bq : (z == 1) ? bk : bv;

  stage_panel_async(hb + (size_t)tm * 16 * Cc, As, threadIdx.x);

  int cb = cg * 256 + wid * 64;                // this wave's 64 columns
  v8f acc[4] = {};
#pragma unroll 2
  for (int kt = 0; kt < Cc / 32; ++kt) {
    const bf16_t* brow = W + (size_t)(kt * 32) * Cc + cb;
    v16bf b0 = load_b_row(brow +  0, Cc);      // issue all VMEM first,
    v16bf b1 = load_b_row(brow + 16, Cc);      // then WMMAs -> partial waits
    v16bf b2 = load_b_row(brow + 32, Cc);
    v16bf b3 = load_b_row(brow + 48, Cc);
    v16bf a  = load_a_row(As + kt * 32, Cc);   // from LDS
    acc[0] = wmma_bf16(a, b0, acc[0]);
    acc[1] = wmma_bf16(a, b1, acc[1]);
    acc[2] = wmma_bf16(a, b2, acc[2]);
    acc[3] = wmma_bf16(a, b3, acc[3]);
  }
  int head = cb >> 6;                          // one head per wave (cb mult of 64)
  bf16_t* outz = qkv + (size_t)z * BHc * Nc * Dc;
  int rbase = tm * 16 + ((lane >> 4) << 3);
#pragma unroll
  for (int t = 0; t < 4; ++t) {
    int cn = cb + t * 16 + (lane & 15);
    int d  = cn & 63;
    float bia = bias[cn];
#pragma unroll
    for (int j = 0; j < 8; ++j) {
      int gn = rbase + j;
      int b_ = gn >> 14, n = gn & (Nc - 1);
      outz[((size_t)(b_ * Hc + head) * Nc + n) * Dc + d] = (bf16_t)(acc[t][j] + bia);
    }
  }
}

// ---------------------------------------------------------------------------
// K2: phi(x) = relu(x @ proj^T) + eps.
//   z=0: qp [BH,N,MP] row-major ; z=1: kpT [BH,MP,N] (transposed store).
// grid (4608, BH, 2), block 128.
// ---------------------------------------------------------------------------
__global__ void k_phi(const bf16_t* __restrict__ qkv, const bf16_t* __restrict__ projb,
                      bf16_t* __restrict__ qp, bf16_t* __restrict__ kpT) {
  int lane = threadIdx.x & 31, wid = threadIdx.x >> 5;
  int bh = blockIdx.y, z = blockIdx.z;
  int tile = blockIdx.x * 4 + wid;             // 1024*18 tiles
  int tm = tile / 18, tc = tile % 18;
  const bf16_t* src = qkv + (size_t)z * BHc * Nc * Dc + (size_t)bh * Nc * Dc
                          + (size_t)tm * 16 * Dc;
  v16bf a0 = load_a_row(src,      Dc);
  v16bf a1 = load_a_row(src + 32, Dc);
  v16bf b0 = load_b_row(projb + tc * 16,                     MPc);
  v16bf b1 = load_b_row(projb + (size_t)32 * MPc + tc * 16,  MPc);
  v8f acc = {};
  acc = wmma_bf16(a0, b0, acc);
  acc = wmma_bf16(a1, b1, acc);

  int m = tc * 16 + (lane & 15);
  int rbase = tm * 16 + ((lane >> 4) << 3);
  if (z == 0) {
    bf16_t* dst = qp + (size_t)bh * Nc * MPc;
#pragma unroll
    for (int j = 0; j < 8; ++j) {
      float v = (m < Mc) ? (fmaxf(acc[j], 0.0f) + KERNEL_EPS) : 0.0f;
      dst[(size_t)(rbase + j) * MPc + m] = (bf16_t)v;
    }
  } else {
    bf16_t* dst = kpT + (size_t)bh * MPc * Nc + (size_t)m * Nc + rbase;
#pragma unroll
    for (int j = 0; j < 8; ++j) {              // contiguous 8 x bf16 per lane
      float v = (m < Mc) ? (fmaxf(acc[j], 0.0f) + KERNEL_EPS) : 0.0f;
      dst[j] = (bf16_t)v;
    }
  }
}

// ---------------------------------------------------------------------------
// K3: ksum[bh,m] = sum_n kpT[bh,m,n].  One wave per (bh,m), coalesced rows.
// grid (BH*MP/8 = 576), block 256.
// ---------------------------------------------------------------------------
__global__ void k_ksum(const bf16_t* __restrict__ kpT, float* __restrict__ ksum) {
  int lane = threadIdx.x & 31, wid = threadIdx.x >> 5;
  int gw = blockIdx.x * 8 + wid;               // over BH*MP
  const bf16_t* row = kpT + (size_t)gw * Nc;
  float s = 0.f;
  for (int n = lane; n < Nc; n += 32) s += (float)row[n];
#pragma unroll
  for (int o = 16; o > 0; o >>= 1) s += __shfl_xor(s, o, 32);
  if (lane == 0) ksum[gw] = s;
}

// ---------------------------------------------------------------------------
// K4: context[bh,m,d] = sum_n kpT[bh,m,n] * v[bh,n,d].
// A = kpT row-major (lda=N).  Wave: 16 m-rows x 64 d-cols (4 accs, A shared).
// grid (6, BH), block 96 (3 waves -> 18 m-tiles).
// ---------------------------------------------------------------------------
__global__ void k_ctx(const bf16_t* __restrict__ kpT, const bf16_t* __restrict__ qkv,
                      bf16_t* __restrict__ ctx) {
  int lane = threadIdx.x & 31, wid = threadIdx.x >> 5;
  int bh = blockIdx.y;
  int tmm = blockIdx.x * 3 + wid;              // m-tile 0..17
  const bf16_t* abase = kpT + (size_t)bh * MPc * Nc + (size_t)tmm * 16 * Nc;
  const bf16_t* vbh   = qkv + (size_t)2 * BHc * Nc * Dc + (size_t)bh * Nc * Dc;

  v8f acc[4] = {};
#pragma unroll 2
  for (int kt = 0; kt < Nc / 32; ++kt) {
    const bf16_t* brow = vbh + (size_t)(kt * 32) * Dc;
    v16bf b0 = load_b_row(brow +  0, Dc);
    v16bf b1 = load_b_row(brow + 16, Dc);
    v16bf b2 = load_b_row(brow + 32, Dc);
    v16bf b3 = load_b_row(brow + 48, Dc);
    v16bf a  = load_a_row(abase + kt * 32, Nc);
    acc[0] = wmma_bf16(a, b0, acc[0]);
    acc[1] = wmma_bf16(a, b1, acc[1]);
    acc[2] = wmma_bf16(a, b2, acc[2]);
    acc[3] = wmma_bf16(a, b3, acc[3]);
  }
  int mbase = tmm * 16 + ((lane >> 4) << 3);
#pragma unroll
  for (int t = 0; t < 4; ++t) {
    int d = t * 16 + (lane & 15);
#pragma unroll
    for (int j = 0; j < 8; ++j)
      ctx[(size_t)bh * MPc * Dc + (size_t)(mbase + j) * Dc + d] = (bf16_t)acc[t][j];
  }
}

// ---------------------------------------------------------------------------
// K5a: d_inv[bh,n] = 1 / dot(qp[bh,n,:], ksum[bh,:]) ; one wave per row.
// ---------------------------------------------------------------------------
__global__ void k_dinv(const bf16_t* __restrict__ qp, const float* __restrict__ ksum,
                       float* __restrict__ dinv) {
  int lane = threadIdx.x & 31, wid = threadIdx.x >> 5;
  int gw = blockIdx.x * 8 + wid;               // over BH*N
  int bh = gw >> 14;
  const bf16_t* row = qp + (size_t)gw * MPc;
  const float*  ks  = ksum + (size_t)bh * MPc;
  float s = 0.f;
  for (int m = lane; m < MPc; m += 32) s += (float)row[m] * ks[m];
#pragma unroll
  for (int o = 16; o > 0; o >>= 1) s += __shfl_xor(s, o, 32);
  if (lane == 0) dinv[gw] = 1.0f / s;
}

// ---------------------------------------------------------------------------
// K5: out[bh,n,d] = d_inv[n] * sum_m qp[n,m]*ctx[m,d] -> attn [B,N,C] bf16.
// Wave: 16 n-rows x 64 d-cols (4 accs share A).  grid (256, BH), block 128.
// ---------------------------------------------------------------------------
__global__ void k_out(const bf16_t* __restrict__ qp, const bf16_t* __restrict__ ctx,
                      const float* __restrict__ dinv, bf16_t* __restrict__ attn) {
  int lane = threadIdx.x & 31, wid = threadIdx.x >> 5;
  int bh = blockIdx.y;
  int tm = blockIdx.x * 4 + wid;               // n-tile 0..1023
  const bf16_t* abase = qp + (size_t)bh * Nc * MPc + (size_t)tm * 16 * MPc;
  const bf16_t* cbh   = ctx + (size_t)bh * MPc * Dc;

  v8f acc[4] = {};
#pragma unroll
  for (int kt = 0; kt < MPc / 32; ++kt) {
    const bf16_t* brow = cbh + (size_t)(kt * 32) * Dc;
    v16bf b0 = load_b_row(brow +  0, Dc);
    v16bf b1 = load_b_row(brow + 16, Dc);
    v16bf b2 = load_b_row(brow + 32, Dc);
    v16bf b3 = load_b_row(brow + 48, Dc);
    v16bf a  = load_a_row(abase + kt * 32, MPc);
    acc[0] = wmma_bf16(a, b0, acc[0]);
    acc[1] = wmma_bf16(a, b1, acc[1]);
    acc[2] = wmma_bf16(a, b2, acc[2]);
    acc[3] = wmma_bf16(a, b3, acc[3]);
  }
  int b_ = bh >> 3, h_ = bh & 7;
  int rbase = tm * 16 + ((lane >> 4) << 3);
#pragma unroll
  for (int j = 0; j < 8; ++j) {
    int n = rbase + j;
    float sc = dinv[(size_t)bh * Nc + n];
#pragma unroll
    for (int t = 0; t < 4; ++t) {
      int d = t * 16 + (lane & 15);
      attn[((size_t)b_ * Nc + n) * Cc + h_ * Dc + d] = (bf16_t)(acc[t][j] * sc);
    }
  }
}

// ---------------------------------------------------------------------------
// K6: pre-LN = attn @ Wo + bo (f32).  Same LDS-staged pattern as k_qkv.
// grid (4096), block 128.
// ---------------------------------------------------------------------------
__global__ void k_proj_o(const bf16_t* __restrict__ attn, const bf16_t* __restrict__ wo,
                         const float* __restrict__ bo, float* __restrict__ outp) {
  __shared__ bf16_t As[16 * Cc];
  int lane = threadIdx.x & 31, wid = threadIdx.x >> 5;
  int tm = blockIdx.x >> 1;
  int cg = blockIdx.x & 1;

  stage_panel_async(attn + (size_t)tm * 16 * Cc, As, threadIdx.x);

  int cb = cg * 256 + wid * 64;
  v8f acc[4] = {};
#pragma unroll 2
  for (int kt = 0; kt < Cc / 32; ++kt) {
    const bf16_t* brow = wo + (size_t)(kt * 32) * Cc + cb;
    v16bf b0 = load_b_row(brow +  0, Cc);
    v16bf b1 = load_b_row(brow + 16, Cc);
    v16bf b2 = load_b_row(brow + 32, Cc);
    v16bf b3 = load_b_row(brow + 48, Cc);
    v16bf a  = load_a_row(As + kt * 32, Cc);
    acc[0] = wmma_bf16(a, b0, acc[0]);
    acc[1] = wmma_bf16(a, b1, acc[1]);
    acc[2] = wmma_bf16(a, b2, acc[2]);
    acc[3] = wmma_bf16(a, b3, acc[3]);
  }
  int rbase = tm * 16 + ((lane >> 4) << 3);
#pragma unroll
  for (int t = 0; t < 4; ++t) {
    int cn = cb + t * 16 + (lane & 15);
    float bia = bo[cn];
#pragma unroll
    for (int j = 0; j < 8; ++j)
      outp[(size_t)(rbase + j) * Cc + cn] = acc[t][j] + bia;
  }
}

// ---------------------------------------------------------------------------
// K7: LayerNorm over C=512 per row; one wave per row. grid 4096, block 256.
// ---------------------------------------------------------------------------
__global__ void k_ln(const float* __restrict__ outp, const float* __restrict__ g,
                     const float* __restrict__ b, float* __restrict__ out) {
  int lane = threadIdx.x & 31, wid = threadIdx.x >> 5;
  size_t row = (size_t)blockIdx.x * 8 + wid;
  const float* x = outp + row * Cc;
  float s = 0.f, s2 = 0.f;
  for (int c = lane; c < Cc; c += 32) { float v = x[c]; s += v; s2 += v * v; }
#pragma unroll
  for (int o = 16; o > 0; o >>= 1) { s += __shfl_xor(s, o, 32); s2 += __shfl_xor(s2, o, 32); }
  float mean = s * (1.0f / Cc);
  float var  = s2 * (1.0f / Cc) - mean * mean;
  float inv  = rsqrtf(var + LN_EPS);
  for (int c = lane; c < Cc; c += 32)
    out[row * Cc + c] = (x[c] - mean) * inv * g[c] + b[c];
}

// ---------------------------------------------------------------------------
// Launch. Workspace (~420 MB with aliasing):
//   hb(32M)[->attn] | wb(2M) | projb(64K) | qkv(96M)[->outp f32 64M] |
//   qp(144M) | kpT(144M) | ksum | ctx | dinv
// ---------------------------------------------------------------------------
extern "C" void kernel_launch(void* const* d_in, const int* in_sizes, int n_in,
                              void* d_out, int out_size, void* d_ws, size_t ws_size,
                              hipStream_t stream) {
  const float* h    = (const float*)d_in[0];
  const float* Wq   = (const float*)d_in[2];
  const float* bq   = (const float*)d_in[3];
  const float* Wk   = (const float*)d_in[4];
  const float* bk   = (const float*)d_in[5];
  const float* Wv   = (const float*)d_in[6];
  const float* bv   = (const float*)d_in[7];
  const float* Wo   = (const float*)d_in[8];
  const float* bo   = (const float*)d_in[9];
  const float* proj = (const float*)d_in[10];
  const float* lng  = (const float*)d_in[11];
  const float* lnb  = (const float*)d_in[12];
  float* out = (float*)d_out;
  char* ws = (char*)d_ws;

  const size_t off_hb    = 0;
  const size_t off_w     = off_hb    + (size_t)Bc * Nc * Cc * 2;        // 33,554,432
  const size_t off_projb = off_w     + 4 * (size_t)Cc * Cc * 2;         // +2,097,152
  const size_t off_qkv   = off_projb + 65536;
  const size_t off_qp    = off_qkv   + 3 * (size_t)BHc * Nc * Dc * 2;   // +100,663,296
  const size_t off_kpT   = off_qp    + (size_t)BHc * Nc * MPc * 2;      // +150,994,944
  const size_t off_ksum  = off_kpT   + (size_t)BHc * Nc * MPc * 2;
  const size_t off_ctx   = off_ksum  + 65536;
  const size_t off_dinv  = off_ctx   + 1048576;

  bf16_t* hb    = (bf16_t*)(ws + off_hb);
  bf16_t* wb    = (bf16_t*)(ws + off_w);
  bf16_t* projb = (bf16_t*)(ws + off_projb);
  bf16_t* qkv   = (bf16_t*)(ws + off_qkv);
  bf16_t* qp    = (bf16_t*)(ws + off_qp);
  bf16_t* kpT   = (bf16_t*)(ws + off_kpT);
  float*  ksum  = (float*) (ws + off_ksum);
  bf16_t* ctx   = (bf16_t*)(ws + off_ctx);
  float*  dinv  = (float*) (ws + off_dinv);
  bf16_t* attn  = (bf16_t*)(ws + off_hb);   // alias: hb dead after K1
  float*  outp  = (float*) (ws + off_qkv);  // alias: q/k/v dead after K4

  k_pack<<<(unsigned)((PK_TOTAL + 255) / 256), 256, 0, stream>>>(
      h, Wq, Wk, Wv, Wo, proj, hb, wb, projb);
  k_qkv<<<dim3(4096, 1, 3), 128, 0, stream>>>(hb, wb, bq, bk, bv, qkv);
  k_phi<<<dim3(4608, BHc, 2), 128, 0, stream>>>(qkv, projb, qp, kpT);
  k_ksum<<<dim3(BHc * MPc / 8), 256, 0, stream>>>(kpT, ksum);
  k_ctx<<<dim3(6, BHc), 96, 0, stream>>>(kpT, qkv, ctx);
  k_dinv<<<dim3(BHc * Nc / 8), 256, 0, stream>>>(qp, ksum, dinv);
  k_out<<<dim3(256, BHc), 128, 0, stream>>>(qp, ctx, dinv, attn);
  k_proj_o<<<dim3(4096), 128, 0, stream>>>(attn, wb + 3 * (size_t)Cc * Cc, bo, outp);
  k_ln<<<dim3((unsigned)(Bc * Nc / 8)), 256, 0, stream>>>(outp, lng, lnb, out);
}
